// Adapter_low_89730456748184
// MI455X (gfx1250) — matrix operand order repaired and checked by reading, not compiled, not run on hardware
//
#include <hip/hip_runtime.h>

typedef __attribute__((ext_vector_type(16))) _Float16 v16h;
typedef __attribute__((ext_vector_type(8)))  _Float16 v8h;
typedef __attribute__((ext_vector_type(4)))  _Float16 v4h;
typedef __attribute__((ext_vector_type(8)))  float    v8f;
typedef __attribute__((ext_vector_type(4)))  unsigned int u32x4;
typedef __attribute__((ext_vector_type(8)))  int      i32x8;
typedef __attribute__((ext_vector_type(4)))  int      i32x4;

#define BB  16
#define HH  64
#define WW  64
#define CC  768
#define DD  192

// ---------------------------------------------------------------- CDNA5 helpers

__device__ __forceinline__ v8f wmma_f16(v16h a, v16h b, v8f c) {
  return __builtin_amdgcn_wmma_f32_16x16x32_f16(false, a, false, b, (short)0, c,
                                                false, false);
}

// async global->LDS copy of 16 bytes; lds_off is a byte offset from the
// workgroup dynamic-LDS base. Tracked with ASYNCcnt (per-wave, in-order).
__device__ __forceinline__ void async_copy_b128(unsigned lds_off, const void* g) {
  asm volatile("global_load_async_to_lds_b128 %0, %1, off"
               :: "v"(lds_off), "v"(g) : "memory");
}
__device__ __forceinline__ void wait_async0() {
  asm volatile("s_wait_asynccnt 0" ::: "memory");
}
__device__ __forceinline__ void wait_async5() {
  asm volatile("s_wait_asynccnt 5" ::: "memory");
}

// A-fragment: 16x32 f16 tile, row-major in LDS, leading dim `ld` (elements).
__device__ __forceinline__ v16h load_a_frag(const _Float16* base, int ld) {
  const int lane = threadIdx.x & 31;
  const int m = lane & 15, hf = lane >> 4;
  const _Float16* row = base + m * ld;
  v8h lo = *(const v8h*)(row + 8 * hf);
  v8h hi = *(const v8h*)(row + 16 + 8 * hf);
  v16h a;
#pragma unroll
  for (int e = 0; e < 8; ++e) { a[e] = lo[e]; a[8 + e] = hi[e]; }
  return a;
}

// A-fragment from an f32 tile in LDS (converted at load time).
__device__ __forceinline__ v16h load_a_frag_f32(const float* base, int ld) {
  const int lane = threadIdx.x & 31;
  const int m = lane & 15, hf = lane >> 4;
  const float* row = base + m * ld;
  float4 q0 = *(const float4*)(row + 8 * hf);
  float4 q1 = *(const float4*)(row + 8 * hf + 4);
  float4 q2 = *(const float4*)(row + 16 + 8 * hf);
  float4 q3 = *(const float4*)(row + 16 + 8 * hf + 4);
  v16h a;
  a[0] = (_Float16)q0.x;  a[1] = (_Float16)q0.y;
  a[2] = (_Float16)q0.z;  a[3] = (_Float16)q0.w;
  a[4] = (_Float16)q1.x;  a[5] = (_Float16)q1.y;
  a[6] = (_Float16)q1.z;  a[7] = (_Float16)q1.w;
  a[8] = (_Float16)q2.x;  a[9] = (_Float16)q2.y;
  a[10] = (_Float16)q2.z; a[11] = (_Float16)q2.w;
  a[12] = (_Float16)q3.x; a[13] = (_Float16)q3.y;
  a[14] = (_Float16)q3.z; a[15] = (_Float16)q3.w;
  return a;
}

// B-fragment: logical B is 32x16 (KxN); storage transposed [n][k], ld elements.
__device__ __forceinline__ v16h load_bT_frag(const _Float16* baseT, int ld) {
  const int lane = threadIdx.x & 31;
  const int n = lane & 15, hf = lane >> 4;
  const _Float16* row = baseT + n * ld;
  v8h lo = *(const v8h*)(row + 8 * hf);
  v8h hi = *(const v8h*)(row + 16 + 8 * hf);
  v16h b;
#pragma unroll
  for (int e = 0; e < 8; ++e) { b[e] = lo[e]; b[8 + e] = hi[e]; }
  return b;
}

__device__ __forceinline__ v8f zero8() {
  v8f z = {0.f, 0.f, 0.f, 0.f, 0.f, 0.f, 0.f, 0.f};
  return z;
}

// ---------------------------------------------------------------- init kernels

__global__ void __launch_bounds__(256) cvt_f32_f16(const float* __restrict__ s,
                                                   _Float16* __restrict__ d, int n) {
  int i = blockIdx.x * 256 + threadIdx.x;
  if (i < n) d[i] = (_Float16)s[i];
}

// a[t] = (1/64) sum_{m=-16}^{15} exp(2*pi*i*m*t/64); circulant A[i][j]=a[(i-j)&63]
__global__ void __launch_bounds__(256) build_dft(_Float16* __restrict__ Ar,
                                                 _Float16* __restrict__ Ai,
                                                 _Float16* __restrict__ AiN) {
  int idx = blockIdx.x * 256 + threadIdx.x;
  if (idx >= 64 * 64) return;
  int i = idx >> 6, j = idx & 63;
  int t = (i - j) & 63;
  float sr = 0.f, si = 0.f;
#pragma unroll 1
  for (int m = -16; m < 16; ++m) {
    int mt = (((m * t) % 64) + 64) & 63;
    float ang = 0.09817477042468103871f * (float)mt;  // 2*pi/64
    float sn, cs;
    sincosf(ang, &sn, &cs);
    sr += cs; si += sn;
  }
  sr *= (1.f / 64.f); si *= (1.f / 64.f);
  Ar[idx]  = (_Float16)sr;
  Ai[idx]  = (_Float16)si;
  AiN[idx] = (_Float16)(-si);
}

// ------------------------------------------------- K1: GEMM1 + bias + GELU
// xs[m,d] = gelu(sum_c x[m,c] * W1[d,c] + b1[d]); M=65536, K=768, N=192.
// 128x64 block tile, BK=32. Both slabs staged HBM->LDS by the async engine
// (A as raw f32, converted at fragment load), distance-2 pipeline, 3 buffers.
// K-loop unrolled by 3 so every LDS pointer is smem+constant (keeps ds_load,
// not flat). Each wave issues exactly 5 async ops per K-batch (4xA + 1xB);
// async loads retire in order, so s_wait_asynccnt 5 retires the batch in use.

#define LDAF 36   // f32 elements
#define LDB1 40   // f16 elements

__global__ void __launch_bounds__(256) gemm1_gelu(const float* __restrict__ x,
                                                  const _Float16* __restrict__ W1h,
                                                  const float* __restrict__ b1,
                                                  _Float16* __restrict__ xsh) {
  extern __shared__ char smem[];
  float* A0 = (float*)smem;                // 3 x 128*36 f32
  float* A1 = A0 + 128 * LDAF;
  float* A2 = A1 + 128 * LDAF;
  _Float16* B0 = (_Float16*)(A2 + 128 * LDAF);  // 3 x 64*40 f16
  _Float16* B1 = B0 + 64 * LDB1;
  _Float16* B2 = B1 + 64 * LDB1;

  const int tid = threadIdx.x;
  const int nb = blockIdx.x % 3;
  const int mb = blockIdx.x / 3;
  const size_t m0 = (size_t)mb * 128;
  const int n0 = nb * 64;
  const int wave = tid >> 5;
  const int wm = wave >> 1;
  const int wn = wave & 1;

  v8f acc[2][2];
#pragma unroll
  for (int a = 0; a < 2; ++a)
#pragma unroll
    for (int b = 0; b < 2; ++b) acc[a][b] = zero8();

  auto ISSUE = [&](int kidx, float* Ab, _Float16* Bb) {
    int k0 = kidx * 32;
#pragma unroll
    for (int i = 0; i < 4; ++i) {                       // A: 128x32 f32
      int idx = tid + i * 256;
      int r = idx >> 3, c4 = idx & 7;
      async_copy_b128((unsigned)((char*)(Ab + r * LDAF + c4 * 4) - smem),
                      (const void*)(x + (m0 + r) * CC + k0 + c4 * 4));
    }
    {                                                   // B: 64x32 f16 (W1 rows)
      int n = tid >> 2, seg = tid & 3;
      async_copy_b128((unsigned)((char*)(Bb + n * LDB1 + seg * 8) - smem),
                      (const void*)(W1h + (size_t)(n0 + n) * CC + k0 + seg * 8));
    }
  };

  auto STEP = [&](int it, const float* Ac, const _Float16* Bc,
                  float* Ap, _Float16* Bp) {
    if (it + 1 < 24) wait_async5();   // batch `it` retired; `it+1` may fly
    else             wait_async0();
    __syncthreads();
    if (it + 2 < 24) ISSUE(it + 2, Ap, Bp);

    v16h bf0 = load_bT_frag(Bc + (wn * 32) * LDB1, LDB1);
    v16h bf1 = load_bT_frag(Bc + (wn * 32 + 16) * LDB1, LDB1);
#pragma unroll
    for (int sm = 0; sm < 2; ++sm) {
      v16h af = load_a_frag_f32(Ac + (wm * 32 + sm * 16) * LDAF, LDAF);
      acc[sm][0] = wmma_f16(af, bf0, acc[sm][0]);
      acc[sm][1] = wmma_f16(af, bf1, acc[sm][1]);
    }
  };

  ISSUE(0, A0, B0);
  ISSUE(1, A1, B1);

#pragma unroll 1
  for (int base = 0; base < 24; base += 3) {
    STEP(base + 0, A0, B0, A2, B2);
    STEP(base + 1, A1, B1, A0, B0);
    STEP(base + 2, A2, B2, A1, B1);
  }

  const int lane = tid & 31, hf = lane >> 4, col = lane & 15;
#pragma unroll
  for (int sm = 0; sm < 2; ++sm)
#pragma unroll
    for (int sn = 0; sn < 2; ++sn) {
      int d = n0 + wn * 32 + sn * 16 + col;
      float bias = b1[d];
#pragma unroll
      for (int r = 0; r < 8; ++r) {
        size_t m = m0 + wm * 32 + sm * 16 + r + 8 * hf;
        float v = acc[sm][sn][r] + bias;
        v = 0.5f * v * (1.f + erff(v * 0.70710678118654752440f));  // exact GELU
        xsh[m * DD + d] = (_Float16)v;
      }
    }
}

// ------------------------------------------------- K2: h-axis circular filter
// per (b,w): Tr[h,d] = sum_h' Ar[h,h'] xs[b,h',w,d]; Ti likewise with Ai.

#define LDX 72
#define LDAC 72

__global__ void __launch_bounds__(256) filter_h(const _Float16* __restrict__ xsh,
                                                const _Float16* __restrict__ Arw,
                                                const _Float16* __restrict__ Aiw,
                                                _Float16* __restrict__ Trh,
                                                _Float16* __restrict__ Tih) {
  extern __shared__ char smem[];
  _Float16* XsT = (_Float16*)smem;            // [d=192][h'=64] transposed
  _Float16* Ars = XsT + DD * LDX;
  _Float16* Ais = Ars + 64 * LDAC;
  const int tid = threadIdx.x;
  const int b = blockIdx.x >> 6, w = blockIdx.x & 63;

#pragma unroll
  for (int i = 0; i < 2; ++i) {
    int idx = tid + i * 256;
    int r = idx >> 3, seg = idx & 7;
    async_copy_b128((unsigned)((char*)(Ars + r * LDAC + seg * 8) - smem),
                    (const void*)(Arw + r * 64 + seg * 8));
    async_copy_b128((unsigned)((char*)(Ais + r * LDAC + seg * 8) - smem),
                    (const void*)(Aiw + r * 64 + seg * 8));
  }
#pragma unroll
  for (int i = 0; i < 6; ++i) {
    int idx = tid + i * 256;
    int h = idx / 24, seg = idx % 24;
    v8h v = *(const v8h*)(xsh + ((size_t)((b * 64 + h) * 64 + w)) * DD + seg * 8);
#pragma unroll
    for (int e = 0; e < 8; ++e) XsT[(seg * 8 + e) * LDX + h] = v[e];
  }
  wait_async0();
  __syncthreads();

  const int wave = tid >> 5;
  const int mt = wave & 3;
  const _Float16* Amat = (wave < 4) ? Ars : Ais;
  _Float16* outT = (wave < 4) ? Trh : Tih;
  const int lane = tid & 31, hf = lane >> 4, col = lane & 15;

  v16h a0 = load_a_frag(Amat + mt * 16 * LDAC, LDAC);
  v16h a1 = load_a_frag(Amat + mt * 16 * LDAC + 32, LDAC);
#pragma unroll
  for (int nt = 0; nt < 12; ++nt) {
    v8f c = zero8();
    v16h b0 = load_bT_frag(XsT + nt * 16 * LDX, LDX);
    v16h b1f = load_bT_frag(XsT + nt * 16 * LDX + 32, LDX);
    c = wmma_f16(a0, b0, c);
    c = wmma_f16(a1, b1f, c);
    int d = nt * 16 + col;
#pragma unroll
    for (int r = 0; r < 8; ++r) {
      int hrow = mt * 16 + r + 8 * hf;
      outT[((size_t)((b * 64 + hrow) * 64 + w)) * DD + d] = (_Float16)c[r];
    }
  }
}

// ----------------------------- K3: w-axis filter + abs + GEMM2 + bias + skip
// per (b,h):  Y[w,d] = | Ar Tr - Ai Ti |  ;  out[w,c] = Y @ W2^T + b2 + x
// W2 slabs (contiguous 128x192 f16 = 48KB) streamed via the Tensor Data Mover.

#define LDT 72
#define LDY 200
#define LDW2 192

__global__ void __launch_bounds__(256) filter_w_gemm2(
    const _Float16* __restrict__ Trh, const _Float16* __restrict__ Tih,
    const _Float16* __restrict__ Arw, const _Float16* __restrict__ AiNw,
    const _Float16* __restrict__ W2h, const float* __restrict__ b2,
    const float* __restrict__ x, float* __restrict__ out) {
  extern __shared__ char smem[];
  _Float16* TrT = (_Float16*)smem;            // [d=192][w'=64]
  _Float16* TiT = TrT + DD * LDT;
  _Float16* Ars = TiT + DD * LDT;
  _Float16* Ais = Ars + 64 * LDAC;            // holds -Ai
  _Float16* Ys  = Ais + 64 * LDAC;            // [w=64][d=192] ld 200
  _Float16* W2s = TrT;                        // phase2: 128 x 192 slab (reuse)

  const int tid = threadIdx.x;
  const int b = blockIdx.x >> 6, h = blockIdx.x & 63;
  const _Float16* trsl = Trh + ((size_t)(b * 64 + h)) * 64 * DD;
  const _Float16* tisl = Tih + ((size_t)(b * 64 + h)) * 64 * DD;

#pragma unroll
  for (int i = 0; i < 2; ++i) {
    int idx = tid + i * 256;
    int r = idx >> 3, seg = idx & 7;
    async_copy_b128((unsigned)((char*)(Ars + r * LDAC + seg * 8) - smem),
                    (const void*)(Arw + r * 64 + seg * 8));
    async_copy_b128((unsigned)((char*)(Ais + r * LDAC + seg * 8) - smem),
                    (const void*)(AiNw + r * 64 + seg * 8));
  }
#pragma unroll
  for (int i = 0; i < 6; ++i) {
    int idx = tid + i * 256;
    int wq = idx / 24, seg = idx % 24;
    v8h vr = *(const v8h*)(trsl + (size_t)wq * DD + seg * 8);
    v8h vi = *(const v8h*)(tisl + (size_t)wq * DD + seg * 8);
#pragma unroll
    for (int e = 0; e < 8; ++e) {
      TrT[(seg * 8 + e) * LDT + wq] = vr[e];
      TiT[(seg * 8 + e) * LDT + wq] = vi[e];
    }
  }
  wait_async0();
  __syncthreads();

  const int wave = tid >> 5;
  const int lane = tid & 31, hf = lane >> 4, col = lane & 15;

  // ---- phase 1: Y = abs(Ar*Tr + (-Ai)*Ti)
  {
    int mt = wave >> 1;
    int ntb = (wave & 1) * 6;
    v16h ar0 = load_a_frag(Ars + mt * 16 * LDAC, LDAC);
    v16h ar1 = load_a_frag(Ars + mt * 16 * LDAC + 32, LDAC);
    v16h ai0 = load_a_frag(Ais + mt * 16 * LDAC, LDAC);
    v16h ai1 = load_a_frag(Ais + mt * 16 * LDAC + 32, LDAC);
#pragma unroll
    for (int q = 0; q < 6; ++q) {
      int nt = ntb + q;
      v8f c = zero8();
      v16h br0 = load_bT_frag(TrT + nt * 16 * LDT, LDT);
      v16h br1 = load_bT_frag(TrT + nt * 16 * LDT + 32, LDT);
      v16h bi0 = load_bT_frag(TiT + nt * 16 * LDT, LDT);
      v16h bi1 = load_bT_frag(TiT + nt * 16 * LDT + 32, LDT);
      c = wmma_f16(ar0, br0, c);
      c = wmma_f16(ar1, br1, c);
      c = wmma_f16(ai0, bi0, c);
      c = wmma_f16(ai1, bi1, c);
#pragma unroll
      for (int r = 0; r < 8; ++r)
        Ys[(mt * 16 + r + 8 * hf) * LDY + nt * 16 + col] = (_Float16)fabsf(c[r]);
    }
  }
  __syncthreads();

  // ---- phase 2: out = Y @ W2^T + b2 + x  (M=64, N=768 in 6 chunks of 128, K=192)
  const float* xrow = x + ((size_t)(b * 64 + h)) * 64 * CC;
  float* orow = out + ((size_t)(b * 64 + h)) * 64 * CC;
  const int mt2 = wave & 3;
  const int ntb2 = (wave >> 2) * 4;

  v16h af[6];
#pragma unroll
  for (int ks = 0; ks < 6; ++ks)
    af[ks] = load_a_frag(Ys + mt2 * 16 * LDY + ks * 32, LDY);

  for (int ch = 0; ch < 6; ++ch) {
    __syncthreads();
#if __has_builtin(__builtin_amdgcn_tensor_load_to_lds)
    if (wave == 0) {
      // Tensor DMA descriptor (D#): 2-D tile 192x128 of 2-byte elements,
      // contiguous source, dest = W2s (byte offset 0 in dynamic LDS).
      unsigned long long ga =
          (unsigned long long)(const void*)(W2h + (size_t)ch * 128 * DD);
      u32x4 g0 = {1u,                                            // count=1
                  (unsigned)((char*)W2s - smem),                 // lds_addr
                  (unsigned)(ga & 0xFFFFFFFFu),
                  (unsigned)((ga >> 32) & 0x1FFFFFFu) | (2u << 30)};  // type=2
      i32x8 g1 = {(int)(1u << 16),            // data_size = 1 -> 2 bytes
                  (int)(192u << 16),          // tensor_dim0 = 192 (lo16)
                  (int)(128u << 16),          // tensor_dim1 = 128 (lo16)
                  (int)(192u << 16),          // tile_dim0 = 192
                  (int)128,                   // tile_dim1 = 128
                  (int)192,                   // tensor_dim0_stride = 192
                  (int)(24576u << 16),        // tensor_dim1_stride lo16
                  0};
      i32x4 g2 = {0, 0, 0, 0};
      i32x4 g3 = {0, 0, 0, 0};
#if defined(__clang_major__) && __clang_major__ >= 23
      i32x8 gx = {0, 0, 0, 0, 0, 0, 0, 0};
      __builtin_amdgcn_tensor_load_to_lds(g0, g1, g2, g3, gx, 0);
#else
      __builtin_amdgcn_tensor_load_to_lds(g0, g1, g2, g3, 0);
#endif
      __builtin_amdgcn_s_wait_tensorcnt(0);
    }
#else
#pragma unroll
    for (int i = 0; i < 12; ++i) {
      int idx = tid + i * 256;
      int r = idx / 24, seg = idx % 24;
      async_copy_b128((unsigned)((char*)(W2s + r * LDW2 + seg * 8) - smem),
                      (const void*)(W2h + (size_t)(ch * 128 + r) * DD + seg * 8));
    }
    wait_async0();
#endif
    __syncthreads();
#pragma unroll
    for (int q = 0; q < 4; ++q) {
      int ntl = ntb2 + q;
      v8f c = zero8();
#pragma unroll
      for (int ks = 0; ks < 6; ++ks) {
        v16h bf = load_bT_frag(W2s + ntl * 16 * LDW2 + ks * 32, LDW2);
        c = wmma_f16(af[ks], bf, c);
      }
      int cc = ch * 128 + ntl * 16 + col;
      float bias = b2[cc];
#pragma unroll
      for (int r = 0; r < 8; ++r) {
        int wrow = mt2 * 16 + r + 8 * hf;
        orow[(size_t)wrow * CC + cc] = c[r] + bias + xrow[(size_t)wrow * CC + cc];
      }
    }
  }
}

// ---------------------------------------------------------------- launch

extern "C" void kernel_launch(void* const* d_in, const int* in_sizes, int n_in,
                              void* d_out, int out_size, void* d_ws, size_t ws_size,
                              hipStream_t stream) {
  const float* x  = (const float*)d_in[0];
  const float* W1 = (const float*)d_in[1];
  const float* b1 = (const float*)d_in[2];
  const float* W2 = (const float*)d_in[3];
  const float* b2 = (const float*)d_in[4];
  float* out = (float*)d_out;
  char* ws = (char*)d_ws;

  _Float16* W1h  = (_Float16*)(ws + 0);          // 192*768*2   = 294912
  _Float16* W2h  = (_Float16*)(ws + 294912);     // 768*192*2   = 294912
  _Float16* Arw  = (_Float16*)(ws + 589824);     // 64*64*2     = 8192
  _Float16* Aiw  = (_Float16*)(ws + 598016);     // 8192
  _Float16* AiNw = (_Float16*)(ws + 606208);     // 8192
  _Float16* xsh  = (_Float16*)(ws + 614400);     // 25165824
  _Float16* Trh  = (_Float16*)(ws + 25780224);   // 25165824
  _Float16* Tih  = (_Float16*)(ws + 50946048);   // 25165824 (total 76111872 B)

  const int gemm1_lds = 3 * (128 * LDAF * 4) + 3 * (64 * LDB1 * 2);  // 70656
  hipFuncSetAttribute(reinterpret_cast<const void*>(gemm1_gelu),
                      hipFuncAttributeMaxDynamicSharedMemorySize, gemm1_lds);
  hipFuncSetAttribute(reinterpret_cast<const void*>(filter_w_gemm2),
                      hipFuncAttributeMaxDynamicSharedMemorySize, 99328);
  hipFuncSetAttribute(reinterpret_cast<const void*>(filter_h),
                      hipFuncAttributeMaxDynamicSharedMemorySize, 46080);

  cvt_f32_f16<<<576, 256, 0, stream>>>(W1, W1h, DD * CC);
  cvt_f32_f16<<<576, 256, 0, stream>>>(W2, W2h, CC * DD);
  build_dft<<<16, 256, 0, stream>>>(Arw, Aiw, AiNw);

  gemm1_gelu<<<1536, 256, gemm1_lds, stream>>>(x, W1h, b1, xsh);

  filter_h<<<BB * WW, 256, (DD * LDX + 2 * 64 * LDAC) * 2, stream>>>(
      xsh, Arw, Aiw, Trh, Tih);

  filter_w_gemm2<<<BB * HH, 256,
                   (2 * DD * LDT + 2 * 64 * LDAC + 64 * LDY) * 2, stream>>>(
      Trh, Tih, Arw, AiNw, W2h, b2, x, out);
}
